// Model_73194832658438
// MI455X (gfx1250) — compile-verified, matrix operand order checked by reading, not run on hardware
//
#include <hip/hip_runtime.h>
#include <math.h>
#include <cmath>

// ---------------------------------------------------------------------------
// Autoformer forward for MI455X (gfx1250, wave32, WMMA).
// All matmuls run through v_wmma_f32_16x16x32_bf16 (f32 accumulate).
// FFT autocorrelation is replaced by the mathematically identical
// gram-matrix + wrapped-diagonal-mean formulation.
// ---------------------------------------------------------------------------

typedef __attribute__((ext_vector_type(16))) __bf16 v16bf;
typedef __attribute__((ext_vector_type(4)))  __bf16 v4bf;
typedef __attribute__((ext_vector_type(8)))  float  v8f;
typedef __attribute__((ext_vector_type(4)))  float  vf4;

#define TM 128
#define TN 64
#define TK 64     // K-slab per LDS stage (all call sites have K % 64 == 0)
#define LDP 68    // padded LDS inner dim: 136B rows (8B-aligned stores)

static __device__ inline void cvt_store4(__bf16* dst, vf4 v)
{
    // two v_cvt_pk_bf16_f32 + one ds_store_b64
    *(v4bf*)dst = __builtin_convertvector(v, v4bf);
}

// Generic batched GEMM: C[b] = act(A[b] * B[b] + bias) + resid
//   A: M x K row-major (lda), B: K x N row-major (ldb) or, if transB,
//   B is N x K row-major accessed as its transpose (gram path).
// Requirement: K % 64 == 0 (true for every call site: K in {512, 2048}).
__global__ __launch_bounds__(256)
void gemm_wmma(const float* __restrict__ A, const float* __restrict__ Bm,
               const float* __restrict__ bias, const float* __restrict__ resid,
               float* __restrict__ Cm,
               int M, int N, int K,
               long sA, long sB, long sC,
               int lda, int ldb, int ldc,
               int transB, int doGelu)
{
    __shared__ __bf16 As[TM * LDP];
    __shared__ __bf16 Bs[TN * LDP];

    const int bz = blockIdx.z;
    A  += (long)bz * sA;
    Bm += (long)bz * sB;
    Cm += (long)bz * sC;
    const float* Rm = resid ? resid + (long)bz * sC : nullptr;

    const int tid   = threadIdx.x;
    const int wave  = tid >> 5;
    const int lane  = tid & 31;
    const int waveM = wave >> 1;   // 0..3  (rows of waves)
    const int waveN = wave & 1;    // 0..1  (cols of waves)
    const int m0 = blockIdx.y * TM;
    const int n0 = blockIdx.x * TN;
    const int rc = lane & 15;      // row (A) / col (B) within 16
    const int hv = lane >> 4;      // K-half selector

    // Block-uniform full-tile test: true for every GEMM in this model except
    // the final 512x21 projection (M is always a multiple of 128).
    const bool full = (m0 + TM <= M) && (n0 + TN <= N);

    v8f acc[2][2] = {};

    for (int k0 = 0; k0 < K; k0 += TK) {
        // ---- Stage A tile (128 x 64) as bf16 -------------------------------
        if (full) {
            #pragma unroll
            for (int it = 0; it < (TM * TK / 4) / 256; ++it) {   // 8 iterations
                int g  = tid + it * 256;
                int r  = g >> 4;
                int kk = (g & 15) * 4;
                vf4 v = *(const vf4*)(A + (long)(m0 + r) * lda + k0 + kk);
                cvt_store4(&As[r * LDP + kk], v);
            }
            if (transB) {
                #pragma unroll
                for (int it = 0; it < (TN * TK / 4) / 256; ++it) { // 4 iterations
                    int g  = tid + it * 256;
                    int n  = g >> 4;
                    int kk = (g & 15) * 4;
                    vf4 v = *(const vf4*)(Bm + (long)(n0 + n) * ldb + k0 + kk);
                    cvt_store4(&Bs[n * LDP + kk], v);
                }
            } else {
                #pragma unroll
                for (int it = 0; it < (TN * TK / 4) / 256; ++it) { // 4 iterations
                    int g   = tid + it * 256;
                    int kk  = g >> 4;          // 0..63
                    int n4  = (g & 15) * 4;    // 0..60
                    if ((ldb & 3) == 0) {
                        vf4 v = *(const vf4*)(Bm + (long)(k0 + kk) * ldb + n0 + n4);
                        Bs[(n4 + 0) * LDP + kk] = (__bf16)v.x;
                        Bs[(n4 + 1) * LDP + kk] = (__bf16)v.y;
                        Bs[(n4 + 2) * LDP + kk] = (__bf16)v.z;
                        Bs[(n4 + 3) * LDP + kk] = (__bf16)v.w;
                    } else {
                        #pragma unroll
                        for (int j = 0; j < 4; ++j)
                            Bs[(n4 + j) * LDP + kk] =
                                (__bf16)Bm[(long)(k0 + kk) * ldb + n0 + n4 + j];
                    }
                }
            }
        } else {
            #pragma unroll
            for (int it = 0; it < (TM * TK / 4) / 256; ++it) {
                int g  = tid + it * 256;
                int r  = g >> 4;
                int kk = (g & 15) * 4;
                int gm = m0 + r;
                vf4 v = {0.f, 0.f, 0.f, 0.f};
                if (gm < M) v = *(const vf4*)(A + (long)gm * lda + k0 + kk);
                cvt_store4(&As[r * LDP + kk], v);
            }
            if (transB) {
                #pragma unroll
                for (int it = 0; it < (TN * TK / 4) / 256; ++it) {
                    int g  = tid + it * 256;
                    int n  = g >> 4;
                    int kk = (g & 15) * 4;
                    int gn = n0 + n;
                    vf4 v = {0.f, 0.f, 0.f, 0.f};
                    if (gn < N) v = *(const vf4*)(Bm + (long)gn * ldb + k0 + kk);
                    cvt_store4(&Bs[n * LDP + kk], v);
                }
            } else {
                #pragma unroll
                for (int it = 0; it < (TN * TK / 4) / 256; ++it) {
                    int g   = tid + it * 256;
                    int kk  = g >> 4;
                    int n4  = (g & 15) * 4;
                    int gk  = k0 + kk;
                    int gnb = n0 + n4;
                    #pragma unroll
                    for (int j = 0; j < 4; ++j) {
                        float v = (gnb + j < N) ? Bm[(long)gk * ldb + gnb + j] : 0.0f;
                        Bs[(n4 + j) * LDP + kk] = (__bf16)v;
                    }
                }
            }
        }
        // Prefetch next A K-slab (speculative) -> global_prefetch_b8
        if (k0 + TK < K) {
            int pr = m0 + (tid >> 1);
            if (pr < M)
                __builtin_prefetch(A + (long)pr * lda + k0 + TK + (tid & 1) * 32, 0, 1);
        }
        __syncthreads();

        // ---- 2 K-substeps x 2x2 WMMA tiles = 8 wmma per stage --------------
        #pragma unroll
        for (int ks = 0; ks < TK; ks += 32) {
            #pragma unroll
            for (int mt = 0; mt < 2; ++mt) {
                const int ar = waveM * 32 + mt * 16 + rc;
                v16bf af;
                #pragma unroll
                for (int e = 0; e < 8; ++e) {
                    af[e]     = As[ar * LDP + ks + hv * 8 + e];
                    af[e + 8] = As[ar * LDP + ks + 16 + hv * 8 + e];
                }
                #pragma unroll
                for (int nt = 0; nt < 2; ++nt) {
                    const int bn = waveN * 32 + nt * 16 + rc;
                    v16bf bf;
                    #pragma unroll
                    for (int e = 0; e < 8; ++e) {
                        bf[e]     = Bs[bn * LDP + ks + hv * 8 + e];
                        bf[e + 8] = Bs[bn * LDP + ks + 16 + hv * 8 + e];
                    }
                    acc[mt][nt] = __builtin_amdgcn_wmma_f32_16x16x32_bf16(
                        false, af, false, bf, (short)0, acc[mt][nt], false, false);
                }
            }
        }
        __syncthreads();
    }

    // Store: C/D layout -> lane rc = N, VGPR j -> M = hv*8 + j
    #pragma unroll
    for (int mt = 0; mt < 2; ++mt) {
        #pragma unroll
        for (int nt = 0; nt < 2; ++nt) {
            const int gm0 = m0 + waveM * 32 + mt * 16 + hv * 8;
            const int gn  = n0 + waveN * 32 + nt * 16 + rc;
            #pragma unroll
            for (int j = 0; j < 8; ++j) {
                const int gm = gm0 + j;
                if (full || (gm < M && gn < N)) {
                    float v = acc[mt][nt][j];
                    if (bias)   v += bias[gn];
                    if (doGelu) v = 0.5f * v * (1.0f + erff(v * 0.70710678118654752f));
                    if (Rm)     v += Rm[(long)gm * ldc + gn];
                    Cm[(long)gm * ldc + gn] = v;
                }
            }
        }
    }
}

// series_decomp: moving average (k=25, edge replicate); seas = x - mov
__global__ void decomp_kernel(const float* __restrict__ x,
                              float* __restrict__ seas, float* __restrict__ trend,
                              int Bn, int L, int C, int trendAccum)
{
    long i = (long)blockIdx.x * blockDim.x + threadIdx.x;
    if (i >= (long)Bn * L * C) return;
    int c = (int)(i % C);
    int t = (int)((i / C) % L);
    int b = (int)(i / ((long)C * L));
    const float* xb = x + (long)b * L * C;
    float s = 0.0f;
    for (int j = t - 12; j <= t + 12; ++j) {
        int jj = j < 0 ? 0 : (j >= L ? L - 1 : j);
        s += xb[(long)jj * C + c];
    }
    s *= (1.0f / 25.0f);
    if (seas)  seas[i] = xb[(long)t * C + c] - s;
    if (trend) { if (trendAccum) trend[i] += s; else trend[i] = s; }
}

__global__ void timemean_kernel(const float* __restrict__ x, float* __restrict__ out,
                                int Bn, int L, int C)
{
    int i = blockIdx.x * blockDim.x + threadIdx.x;
    if (i >= Bn * C) return;
    int c = i % C, b = i / C;
    const float* xb = x + (long)b * L * C + c;
    float s = 0.0f;
    for (int t = 0; t < L; ++t) s += xb[(long)t * C];
    out[i] = s / (float)L;
}

__global__ void build_dec_init(const float* __restrict__ seas, const float* __restrict__ trend,
                               const float* __restrict__ meanB,
                               float* __restrict__ seasInit, float* __restrict__ trendInit,
                               int Bn, int L, int LI, int LBL, int C)
{
    long i = (long)blockIdx.x * blockDim.x + threadIdx.x;
    if (i >= (long)Bn * LI * C) return;
    int c = (int)(i % C);
    int t = (int)((i / C) % LI);
    int b = (int)(i / ((long)C * LI));
    if (t < LBL) {
        long src = ((long)b * L + (L - LBL + t)) * C + c;
        seasInit[i]  = seas[src];
        trendInit[i] = trend[src];
    } else {
        seasInit[i]  = 0.0f;
        trendInit[i] = meanB[b * C + c];
    }
}

// circular conv1d, kernel width 3, weights W[Cout][Cin][3]
__global__ void circconv_kernel(const float* __restrict__ x, const float* __restrict__ W,
                                float* __restrict__ out,
                                int Bn, int L, int Cin, int Cout, int accum)
{
    long i = (long)blockIdx.x * blockDim.x + threadIdx.x;
    if (i >= (long)Bn * L * Cout) return;
    int co = (int)(i % Cout);
    int t  = (int)((i / Cout) % L);
    int b  = (int)(i / ((long)Cout * L));
    const float* xb = x + (long)b * L * Cin;
    float s = 0.0f;
    for (int tap = 0; tap < 3; ++tap) {
        int ts = t + tap - 1;
        if (ts < 0) ts += L; else if (ts >= L) ts -= L;
        const float* xr = xb + (long)ts * Cin;
        const float* wr = W + (long)co * Cin * 3 + tap;
        for (int ci = 0; ci < Cin; ++ci) s += xr[ci] * wr[ci * 3];
    }
    if (accum) out[i] += s; else out[i] = s;
}

// mean over (H,E) of circular cross-correlation via gram wrapped diagonals:
// mc[b,n] = inv * sum_t G[b][t, (t-n) mod L]
__global__ __launch_bounds__(256)
void diagmean_kernel(const float* __restrict__ G, float* __restrict__ mc,
                     int L, int mcStride, float inv)
{
    int n = blockIdx.x, b = blockIdx.y;
    const float* Gb = G + (long)b * L * L;
    float s = 0.0f;
    for (int t = threadIdx.x; t < L; t += 256) {
        int j = t - n; if (j < 0) j += L;
        s += Gb[(long)t * L + j];
    }
    __shared__ float red[256];
    red[threadIdx.x] = s; __syncthreads();
    for (int off = 128; off > 0; off >>= 1) {
        if (threadIdx.x < off) red[threadIdx.x] += red[threadIdx.x + off];
        __syncthreads();
    }
    if (threadIdx.x == 0) mc[(long)b * mcStride + n] = red[0] * inv;
}

// per-batch top-k (descending, first-index tie-break) + softmax weights
__global__ void topk_kernel(const float* __restrict__ mc, int mcStride,
                            float* __restrict__ wOut, int* __restrict__ dOut,
                            int L, int KTOP)
{
    int b = blockIdx.x;
    const float* row = mc + (long)b * mcStride;
    __shared__ int   chosen[32];
    __shared__ float vals[32];
    int lane = threadIdx.x;
    for (int i = 0; i < KTOP; ++i) {
        float best = -3.4e38f; int bidx = 0x7fffffff;
        for (int t = lane; t < L; t += 32) {
            bool used = false;
            for (int j = 0; j < i; ++j) if (chosen[j] == t) used = true;
            if (!used) {
                float v = row[t];
                if (v > best || (v == best && t < bidx)) { best = v; bidx = t; }
            }
        }
        for (int off = 16; off > 0; off >>= 1) {
            float ov = __shfl_xor(best, off);
            int   oi = __shfl_xor(bidx, off);
            if (ov > best || (ov == best && oi < bidx)) { best = ov; bidx = oi; }
        }
        if (lane == 0) { chosen[i] = bidx; vals[i] = best; }
        __syncthreads();
    }
    if (lane == 0) {
        float mx = vals[0];
        for (int i = 1; i < KTOP; ++i) mx = fmaxf(mx, vals[i]);
        float e[32], s = 0.0f;
        for (int i = 0; i < KTOP; ++i) { e[i] = expf(vals[i] - mx); s += e[i]; }
        for (int i = 0; i < KTOP; ++i) {
            wOut[b * 32 + i] = e[i] / s;
            dOut[b * 32 + i] = chosen[i];
        }
    }
}

// out[b,t,d] = sum_i w[b,i] * v[b, (t+delay[b,i]) mod L, d]
__global__ void agg_kernel(const float* __restrict__ v, const float* __restrict__ wTop,
                           const int* __restrict__ dTop, float* __restrict__ out,
                           int Bn, int L, int D, long vStride, int KTOP)
{
    long i = (long)blockIdx.x * blockDim.x + threadIdx.x;
    if (i >= (long)Bn * L * D) return;
    int d = (int)(i % D);
    int t = (int)((i / D) % L);
    int b = (int)(i / ((long)D * L));
    const float* vb = v + (long)b * vStride;
    float s = 0.0f;
    for (int k = 0; k < KTOP; ++k) {
        int sh = t + dTop[b * 32 + k];
        if (sh >= L) sh -= L;
        s += wTop[b * 32 + k] * vb[(long)sh * D + d];
    }
    out[i] = s;
}

__global__ __launch_bounds__(256)
void layernorm_kernel(const float* __restrict__ x, const float* __restrict__ g,
                      const float* __restrict__ be, float* __restrict__ out, int D)
{
    int r = blockIdx.x;
    const float* xr = x + (long)r * D;
    float s = 0.0f, s2 = 0.0f;
    for (int c = threadIdx.x; c < D; c += 256) { float v = xr[c]; s += v; s2 += v * v; }
    __shared__ float r1[256], r2[256];
    r1[threadIdx.x] = s; r2[threadIdx.x] = s2; __syncthreads();
    for (int off = 128; off > 0; off >>= 1) {
        if (threadIdx.x < off) { r1[threadIdx.x] += r1[threadIdx.x + off];
                                 r2[threadIdx.x] += r2[threadIdx.x + off]; }
        __syncthreads();
    }
    float mean = r1[0] / (float)D;
    float var  = r2[0] / (float)D - mean * mean;
    float inv  = 1.0f / sqrtf(var + 1e-5f);
    for (int c = threadIdx.x; c < D; c += 256)
        out[(long)r * D + c] = (xr[c] - mean) * inv * g[c] + be[c];
}

__global__ void subcolmean_kernel(float* __restrict__ x, int Bn, int L, int D)
{
    int i = blockIdx.x * blockDim.x + threadIdx.x;
    if (i >= Bn * D) return;
    int d = i % D, b = i / D;
    float* xb = x + (long)b * L * D + d;
    float s = 0.0f;
    for (int t = 0; t < L; ++t) s += xb[(long)t * D];
    s /= (float)L;
    for (int t = 0; t < L; ++t) xb[(long)t * D] -= s;
}

__global__ void final_kernel(const float* __restrict__ trendF, const float* __restrict__ spart,
                             float* __restrict__ out, int Bn, int LI, int PL, int C)
{
    long i = (long)blockIdx.x * blockDim.x + threadIdx.x;
    if (i >= (long)Bn * PL * C) return;
    int c = (int)(i % C);
    int t = (int)((i / C) % PL);
    int b = (int)(i / ((long)C * PL));
    long src = ((long)b * LI + (LI - PL) + t) * C + c;
    out[i] = trendF[src] + spart[src];
}

// ---------------------------------------------------------------------------
extern "C" void kernel_launch(void* const* d_in, const int* in_sizes, int n_in,
                              void* d_out, int out_size, void* d_ws, size_t ws_size,
                              hipStream_t stream)
{
    (void)in_sizes; (void)out_size; (void)ws_size;
    if (n_in < 48) return;

    const int Bn = 16, LE = 512, LD = 384, D = 512, DFF = 2048, C = 21, LBL = 128, PL = 256;
    auto P = [&](int i) { return (const float*)d_in[i]; };

    // Input flattening (setup_inputs() dict insertion order):
    const float* x_enc   = P(0);
    const float* enc_emb = P(1);
    const float* dec_emb = P(2);
    struct AttnW { const float *Wq,*bq,*Wk,*bk,*Wv,*bv,*Wo,*bo; };
    auto getAttn = [&](int base) {
        AttnW a; a.Wq=P(base); a.bq=P(base+1); a.Wk=P(base+2); a.bk=P(base+3);
        a.Wv=P(base+4); a.bv=P(base+5); a.Wo=P(base+6); a.bo=P(base+7); return a;
    };
    AttnW encA[2]        = { getAttn(3), getAttn(13) };
    const float* encW1[2]= { P(11), P(21) };
    const float* encW2[2]= { P(12), P(22) };
    const float* enc_g = P(23); const float* enc_b = P(24);
    AttnW decSelf = getAttn(25), decCross = getAttn(33);
    const float* decW1 = P(41); const float* decW2 = P(42); const float* Wt = P(43);
    const float* dec_g = P(44); const float* dec_b = P(45);
    const float* Wp    = P(46); const float* bp    = P(47);
    float* out = (float*)d_out;

    // Workspace layout (floats)
    float* w = (float*)d_ws;
    size_t off = 0;
    auto alloc = [&](size_t n) { float* p = w + off; off += n; return p; };
    const size_t EA = (size_t)Bn * LE * D;        // 4.19M
    float* bufEnc  = alloc(EA);
    float* bufEncF = alloc(EA);
    float* bufQ    = alloc(EA);
    float* bufK    = alloc(EA);
    float* bufV    = alloc(EA);
    float* bufG    = alloc(EA);
    float* bufT1   = alloc(EA);
    float* bufT2   = alloc(EA);
    float* bufFF   = alloc((size_t)Bn * LE * DFF); // 16.8M
    float* bufDec  = alloc((size_t)Bn * LD * D);
    float* bufTsum = alloc((size_t)Bn * LD * D);
    float* seasS   = alloc((size_t)Bn * LE * C);
    float* trendS  = alloc((size_t)Bn * LE * C);
    float* meanS   = alloc((size_t)Bn * C);
    float* trendF  = alloc((size_t)Bn * LD * C);
    float* seasI   = alloc((size_t)Bn * LD * C);
    float* spart   = alloc((size_t)Bn * LD * C);
    float* mcorr   = alloc((size_t)Bn * 512);
    float* wtop    = alloc((size_t)Bn * 32);
    int*   dtop    = (int*)alloc((size_t)Bn * 32);

    auto launch_gemm = [&](const float* A, const float* Bm, const float* bias,
                           const float* res, float* Cm, int M, int N, int K,
                           long sA, long sB, long sC, int lda, int ldb, int ldc,
                           int trB, int gel, int batch) {
        dim3 g((N + TN - 1) / TN, (M + TM - 1) / TM, batch);
        gemm_wmma<<<g, 256, 0, stream>>>(A, Bm, bias, res, Cm, M, N, K,
                                         sA, sB, sC, lda, ldb, ldc, trB, gel);
    };
    auto decomp = [&](const float* x, float* seas, float* trend, int L, int Cc, int acc) {
        long tot = (long)Bn * L * Cc;
        decomp_kernel<<<(tot + 255) / 256, 256, 0, stream>>>(x, seas, trend, Bn, L, Cc, acc);
    };
    auto ew = [&](long tot) { return dim3((unsigned)((tot + 255) / 256)); };

    // AutoCorrelation attention: outBuf = agg @ Wo + bo + xq
    auto attention = [&](const float* xq, int Lq, const float* xkv, int Skv,
                         const AttnW& A, float* outBuf) {
        int Mq = Bn * Lq, Mk = Bn * Skv, L = Lq;
        launch_gemm(xq,  A.Wq, A.bq, nullptr, bufQ, Mq, D, D, 0,0,0, D,D,D, 0,0, 1);
        launch_gemm(xkv, A.Wk, A.bk, nullptr, bufK, Mk, D, D, 0,0,0, D,D,D, 0,0, 1);
        launch_gemm(xkv, A.Wv, A.bv, nullptr, bufV, Mk, D, D, 0,0,0, D,D,D, 0,0, 1);
        // gram: per-batch Q (LxD) x K^T  (K truncated to first L rows)
        launch_gemm(bufQ, bufK, nullptr, nullptr, bufG, L, L, D,
                    (long)Lq * D, (long)Skv * D, (long)L * L, D, D, L, 1, 0, Bn);
        dim3 dg(L, Bn);
        diagmean_kernel<<<dg, 256, 0, stream>>>(bufG, mcorr, L, 512, 1.0f / 512.0f);
        int KT = (int)(3.0 * log((double)L));
        topk_kernel<<<Bn, 32, 0, stream>>>(mcorr, 512, wtop, dtop, L, KT);
        long tot = (long)Bn * L * D;
        agg_kernel<<<ew(tot), 256, 0, stream>>>(bufV, wtop, dtop, bufT1,
                                                Bn, L, D, (long)Skv * D, KT);
        launch_gemm(bufT1, A.Wo, A.bo, xq, outBuf, Mq, D, D, 0,0,0, D,D,D, 0,0, 1);
    };

    // ---- trend/seasonal init path -------------------------------------------
    timemean_kernel<<<ew(Bn * C), 256, 0, stream>>>(x_enc, meanS, Bn, LE, C);
    decomp(x_enc, seasS, trendS, LE, C, 0);
    build_dec_init<<<ew((long)Bn * LD * C), 256, 0, stream>>>(
        seasS, trendS, meanS, seasI, trendF, Bn, LE, LD, LBL, C);

    // ---- embeddings ---------------------------------------------------------
    circconv_kernel<<<ew((long)Bn * LE * D), 256, 0, stream>>>(
        x_enc, enc_emb, bufEnc, Bn, LE, C, D, 0);
    circconv_kernel<<<ew((long)Bn * LD * D), 256, 0, stream>>>(
        seasI, dec_emb, bufDec, Bn, LD, C, D, 0);

    // ---- encoder ------------------------------------------------------------
    for (int l = 0; l < 2; ++l) {
        attention(bufEnc, LE, bufEnc, LE, encA[l], bufT2);
        decomp(bufT2, bufEnc, nullptr, LE, D, 0);
        launch_gemm(bufEnc, encW1[l], nullptr, nullptr, bufFF,
                    Bn * LE, DFF, D, 0,0,0, D, DFF, DFF, 0, 1, 1);
        launch_gemm(bufFF, encW2[l], nullptr, bufEnc, bufT2,
                    Bn * LE, D, DFF, 0,0,0, DFF, D, D, 0, 0, 1);
        decomp(bufT2, bufEnc, nullptr, LE, D, 0);
    }
    layernorm_kernel<<<Bn * LE, 256, 0, stream>>>(bufEnc, enc_g, enc_b, bufEncF, D);
    subcolmean_kernel<<<ew(Bn * D), 256, 0, stream>>>(bufEncF, Bn, LE, D);

    // ---- decoder ------------------------------------------------------------
    attention(bufDec, LD, bufDec, LD, decSelf, bufT2);
    decomp(bufT2, bufDec, bufTsum, LD, D, 0);
    attention(bufDec, LD, bufEncF, LE, decCross, bufT2);
    decomp(bufT2, bufDec, bufTsum, LD, D, 1);
    launch_gemm(bufDec, decW1, nullptr, nullptr, bufFF,
                Bn * LD, DFF, D, 0,0,0, D, DFF, DFF, 0, 1, 1);
    launch_gemm(bufFF, decW2, nullptr, bufDec, bufT2,
                Bn * LD, D, DFF, 0,0,0, DFF, D, D, 0, 0, 1);
    decomp(bufT2, bufDec, bufTsum, LD, D, 1);
    // residual trend: trendF += circ_conv1d(t1+t2+t3, Wt)
    circconv_kernel<<<ew((long)Bn * LD * C), 256, 0, stream>>>(
        bufTsum, Wt, trendF, Bn, LD, D, C, 1);
    layernorm_kernel<<<Bn * LD, 256, 0, stream>>>(bufDec, dec_g, dec_b, bufT1, D);
    subcolmean_kernel<<<ew(Bn * D), 256, 0, stream>>>(bufT1, Bn, LD, D);
    launch_gemm(bufT1, Wp, bp, nullptr, spart,
                Bn * LD, C, D, 0,0,0, D, C, C, 0, 0, 1);

    // ---- output -------------------------------------------------------------
    final_kernel<<<ew((long)Bn * PL * C), 256, 0, stream>>>(
        trendF, spart, out, Bn, LD, PL, C);
}
